// GCN2_16587163697489
// MI455X (gfx1250) — matrix operand order repaired and checked by reading, not compiled, not run on hardware
//
#include <hip/hip_runtime.h>
#include <hip/hip_bf16.h>

typedef __attribute__((ext_vector_type(16))) __bf16 bf16x16;
typedef __attribute__((ext_vector_type(8)))  float  f32x8;
typedef __attribute__((ext_vector_type(4)))  float  f32x4;

__device__ __forceinline__ unsigned short f32_to_bf16(float f) {
    unsigned int u = __float_as_uint(f);
    u += 0x7fffu + ((u >> 16) & 1u);          // round-to-nearest-even
    return (unsigned short)(u >> 16);
}

// ---------------------------------------------------------------------------
// Degree / normalization:  deg[v] = 1 (self loop) + in-degree(v);  dis = rsqrt
// ---------------------------------------------------------------------------
__global__ void gcn_init_deg(float* __restrict__ deg, int N) {
    int i = blockIdx.x * blockDim.x + threadIdx.x;
    if (i < N) deg[i] = 1.0f;
}

__global__ void gcn_count_deg(const int* __restrict__ cols, float* __restrict__ deg, int E) {
    int e = blockIdx.x * blockDim.x + threadIdx.x;
    if (e < E) unsafeAtomicAdd(&deg[cols[e]], 1.0f);
}

__global__ void gcn_make_dis(float* __restrict__ deg, int N) {
    int i = blockIdx.x * blockDim.x + threadIdx.x;
    if (i < N) deg[i] = rsqrtf(deg[i]);
}

// ---------------------------------------------------------------------------
// W [128 x NOUT] row-major f32  ->  Wt [NOUT x 128] bf16 (column of W contiguous)
// Matches WMMA B-fragment: lane n loads 16 contiguous K values.
// ---------------------------------------------------------------------------
__global__ void gcn_make_wt(const float* __restrict__ W, unsigned short* __restrict__ Wt, int NOUT) {
    int idx = blockIdx.x * blockDim.x + threadIdx.x;
    if (idx >= NOUT * 128) return;
    int n = idx >> 7;          // output channel
    int k = idx & 127;         // input channel
    Wt[idx] = f32_to_bf16(W[(size_t)k * NOUT + n]);
}

// ---------------------------------------------------------------------------
// GEMM:  T = act(A) @ W   (A: [M,128] f32, K=128, NOUT = NT*16)
// Epilogue per node v, channel c:
//     Sout[v][c] = bf16( dis[v] * T[v][c] )            (dis-row-prescaled msgs)
//     Init[v][c] = dis[v]^2 * T[v][c] + bias[c]        (self-loop + bias)
// One wave = one 16-row strip; block = 4 waves = 64 rows.
// ---------------------------------------------------------------------------
template<int NT, bool RELU>
__global__ __launch_bounds__(128) void gcn_gemm(
    const float* __restrict__ A,
    const unsigned short* __restrict__ Wt,   // [NT*16][128] bf16
    const float* __restrict__ bias,          // [NT*16]
    const float* __restrict__ dis,           // [M]
    unsigned short* __restrict__ Sout,       // [M, NT*16] bf16
    float* __restrict__ Init,                // [M, NT*16] f32
    int M)
{
    constexpr int NOUT = NT * 16;
    const int lane    = threadIdx.x & 31;
    const int wid     = threadIdx.x >> 5;
    const int rowBase = blockIdx.x * 64 + wid * 16;
    const int n0      = lane & 15;
    const int hi      = lane >> 4;
    const int mrow    = rowBase + n0;        // A-fragment row for this lane
    const bool mvalid = (mrow < M);

    f32x8 acc[NT] = {};

    const float* arow = A + (size_t)mrow * 128;

#pragma unroll
    for (int kk = 0; kk < 4; ++kk) {         // K = 4 * 32
        bf16x16 afrag;
        if (mvalid) {
            const float* p0 = arow + kk * 32 + hi * 16;
            f32x4 c0 = *(const f32x4*)(p0);
            f32x4 c1 = *(const f32x4*)(p0 + 4);
            f32x4 c2 = *(const f32x4*)(p0 + 8);
            f32x4 c3 = *(const f32x4*)(p0 + 12);
#pragma unroll
            for (int i = 0; i < 4; ++i) {
                float v0 = RELU ? fmaxf(c0[i], 0.0f) : c0[i];
                float v1 = RELU ? fmaxf(c1[i], 0.0f) : c1[i];
                float v2 = RELU ? fmaxf(c2[i], 0.0f) : c2[i];
                float v3 = RELU ? fmaxf(c3[i], 0.0f) : c3[i];
                afrag[i]      = (__bf16)v0;   // K = kk*32 + hi*16 + i
                afrag[4 + i]  = (__bf16)v1;   // K = kk*32 + hi*16 + 4 + i
                afrag[8 + i]  = (__bf16)v2;   // K = kk*32 + hi*16 + 8 + i
                afrag[12 + i] = (__bf16)v3;   // K = kk*32 + hi*16 + 12 + i
            }
        } else {
#pragma unroll
            for (int i = 0; i < 16; ++i) afrag[i] = (__bf16)0.0f;
        }

#pragma unroll
        for (int t = 0; t < NT; ++t) {
            // B fragment: lane holds column n = t*16+n0, K = kk*32 + hi*16 + [0..15]
            const unsigned short* bp =
                Wt + ((size_t)(t * 16 + n0) * 128) + kk * 32 + hi * 16;
            bf16x16 bfrag = *(const bf16x16*)bp;
            acc[t] = __builtin_amdgcn_wmma_f32_16x16x32_bf16(
                false, afrag, false, bfrag, (short)0, acc[t], false, false);
        }
    }

    // Epilogue: C layout -> row = hi*8 + r, col = n0
#pragma unroll
    for (int r = 0; r < 8; ++r) {
        const int gm = rowBase + hi * 8 + r;
        if (gm < M) {
            const float d = dis[gm];
#pragma unroll
            for (int t = 0; t < NT; ++t) {
                const int gn  = t * 16 + n0;
                const float v = acc[t][r];
                const float sv = d * v;
                Sout[(size_t)gm * NOUT + gn] = f32_to_bf16(sv);
                Init[(size_t)gm * NOUT + gn] = d * sv + bias[gn];
            }
        }
    }
}

// ---------------------------------------------------------------------------
// Edge scatter:  Out[col] += dis[col] * S[row]   (S already scaled by dis[row])
// One wave per edge; lane covers 2*WPL channels (C = 64*WPL).
// bf16 gather halves L2 gather traffic; f32 hardware atomics for accumulate.
// ---------------------------------------------------------------------------
template<int WPL>
__global__ __launch_bounds__(256) void gcn_edge_scatter(
    const int* __restrict__ rows, const int* __restrict__ cols,
    const float* __restrict__ dis,
    const unsigned int* __restrict__ S,      // bf16 pairs, row stride C/2 words
    float* __restrict__ Out, int E)
{
    constexpr int C = WPL * 64;
    const int lane = threadIdx.x & 31;
    const long long e = (long long)blockIdx.x * 8 + (threadIdx.x >> 5);
    if (e >= E) return;
    const int r = rows[e];
    const int c = cols[e];
    const float w = dis[c];
    const unsigned int* sp = S + (size_t)r * (C / 2) + lane * WPL;
    float* op = Out + (size_t)c * C + lane * (2 * WPL);
#pragma unroll
    for (int i = 0; i < WPL; ++i) {
        unsigned int v = sp[i];
        unsafeAtomicAdd(op + 2 * i,     w * __uint_as_float(v << 16));
        unsafeAtomicAdd(op + 2 * i + 1, w * __uint_as_float(v & 0xffff0000u));
    }
}

// ---------------------------------------------------------------------------
extern "C" void kernel_launch(void* const* d_in, const int* in_sizes, int n_in,
                              void* d_out, int out_size, void* d_ws, size_t ws_size,
                              hipStream_t stream)
{
    (void)n_in; (void)out_size; (void)ws_size;
    const float* x  = (const float*)d_in[0];
    const int*   ei = (const int*)d_in[1];
    const float* W1 = (const float*)d_in[2];
    const float* b1 = (const float*)d_in[3];
    const float* W2 = (const float*)d_in[4];
    const float* b2 = (const float*)d_in[5];
    float* out = (float*)d_out;

    const int N = in_sizes[0] / 128;
    const int E = in_sizes[1] / 2;
    const int* rows = ei;        // edge_index[0]
    const int* cols = ei + E;    // edge_index[1]

    // workspace carve-up (256B aligned)
    char* ws = (char*)d_ws;
    size_t off = 0;
    auto take = [&](size_t bytes) {
        char* p = ws + off;
        off = (off + bytes + 255) & ~(size_t)255;
        return p;
    };
    float*          dis  = (float*)take((size_t)N * 4);
    unsigned short* w1t  = (unsigned short*)take((size_t)128 * 128 * 2);
    unsigned short* w2t  = (unsigned short*)take((size_t)64 * 128 * 2);
    unsigned short* s1   = (unsigned short*)take((size_t)N * 128 * 2);
    float*          agg1 = (float*)take((size_t)N * 128 * 4);
    unsigned short* s2   = (unsigned short*)take((size_t)N * 64 * 2);

    // 1) normalization: deg -> dis (recomputed every call; graph-capture safe)
    gcn_init_deg <<<(N + 255) / 256, 256, 0, stream>>>(dis, N);
    gcn_count_deg<<<(E + 255) / 256, 256, 0, stream>>>(cols, dis, E);
    gcn_make_dis <<<(N + 255) / 256, 256, 0, stream>>>(dis, N);

    // 2) transpose+convert weights to bf16 for WMMA B fragments
    gcn_make_wt<<<(128 * 128 + 255) / 256, 256, 0, stream>>>(W1, w1t, 128);
    gcn_make_wt<<<(128 * 64  + 255) / 256, 256, 0, stream>>>(W2, w2t, 64);

    const int gemmBlocks = (N + 63) / 64;
    const int edgeBlocks = (E + 7) / 8;

    // 3) layer 1: t1 = X@W1 ; s1 = dis*t1 (bf16) ; agg1 = dis^2*t1 + b1
    gcn_gemm<8, false><<<gemmBlocks, 128, 0, stream>>>(x, w1t, b1, dis, s1, agg1, N);
    //    agg1[col] += dis[col] * s1[row]
    gcn_edge_scatter<2><<<edgeBlocks, 256, 0, stream>>>(rows, cols, dis,
                                                        (const unsigned int*)s1, agg1, E);

    // 4) layer 2: t2 = relu(agg1)@W2 ; s2 = dis*t2 ; out = dis^2*t2 + b2
    gcn_gemm<4, true><<<gemmBlocks, 128, 0, stream>>>(agg1, w2t, b2, dis, s2, out, N);
    //    out[col] += dis[col] * s2[row]
    gcn_edge_scatter<1><<<edgeBlocks, 256, 0, stream>>>(rows, cols, dis,
                                                        (const unsigned int*)s2, out, E);
}
